// ImplicitDecoder_20521353740466
// MI455X (gfx1250) — compile-verified, matrix operand order checked by reading, not run on hardware
//
#include <hip/hip_runtime.h>

// ---------------------------------------------------------------------------
// MI455X / gfx1250 implementation.
//  - wave32; 128-thread blocks = 4 waves.
//  - MLP layers 1 & 2 (75% of FLOPs) run on v_wmma_f32_16x16x32_f16.
//  - Trunk (≈1 G MAC) is scalar VALU; everything fits in 192 MB L2.
// ---------------------------------------------------------------------------

typedef __attribute__((ext_vector_type(16))) _Float16 v16h;
typedef __attribute__((ext_vector_type(8)))  _Float16 v8h;
typedef __attribute__((ext_vector_type(8)))  float    v8f;

#define NP (64 * 64 * 64)   // 262144 query points
#define EPSI 1e-5f

__device__ __forceinline__ float relu_f(float v) { return v > 0.f ? v : 0.f; }

// ---------------------------------------------------------------------------
// Trunk kernels (small volumes, scalar)
// ---------------------------------------------------------------------------

// x = relu(up2(latent, w_up1, b_up1)) : (256) -> (128, 2,2,2)
__global__ void k_up1(const float* __restrict__ lat, const float* __restrict__ w,
                      const float* __restrict__ b, float* __restrict__ out) {
    int idx = blockIdx.x * blockDim.x + threadIdx.x;   // 1024
    if (idx >= 128 * 8) return;
    int o = idx >> 3, s = idx & 7;                     // s = x*4+y*2+z (spatial==kernel pos)
    float acc = b[o];
    for (int i = 0; i < 256; ++i) acc += lat[i] * w[(i * 128 + o) * 8 + s];
    out[idx] = relu_f(acc);
}

// up2: (128,2,2,2) -> (64,4,4,4), bias only (inorm+relu follow)
__global__ void k_up2(const float* __restrict__ x1, const float* __restrict__ w,
                      const float* __restrict__ b, float* __restrict__ out) {
    int idx = blockIdx.x * blockDim.x + threadIdx.x;   // 4096
    if (idx >= 64 * 64) return;
    int o = idx >> 6, sp = idx & 63;
    int D = sp >> 4, H = (sp >> 2) & 3, W = sp & 3;
    int d = D >> 1, x = D & 1, h = H >> 1, y = H & 1, ww = W >> 1, z = W & 1;
    int sIn = d * 4 + h * 2 + ww;
    int sK  = x * 4 + y * 2 + z;
    float acc = b[o];
    for (int i = 0; i < 128; ++i) acc += x1[i * 8 + sIn] * w[(i * 64 + o) * 8 + sK];
    out[idx] = acc;
}

// In-place instance-norm (+ReLU) over S spatial elements; one block per channel.
__global__ void k_inorm_relu(float* __restrict__ buf, int S) {
    int c = blockIdx.x, tid = threadIdx.x;
    __shared__ float ssum[256], ssq[256];
    __shared__ float sm, sr;
    float s = 0.f, q = 0.f;
    float* p = buf + (size_t)c * S;
    for (int i = tid; i < S; i += 256) { float v = p[i]; s += v; q += v * v; }
    ssum[tid] = s; ssq[tid] = q;
    __syncthreads();
    for (int o = 128; o > 0; o >>= 1) {
        if (tid < o) { ssum[tid] += ssum[tid + o]; ssq[tid] += ssq[tid + o]; }
        __syncthreads();
    }
    if (tid == 0) {
        float m = ssum[0] / (float)S;
        float v = ssq[0] / (float)S - m * m;
        sm = m; sr = rsqrtf(v + EPSI);
    }
    __syncthreads();
    for (int i = tid; i < S; i += 256) p[i] = relu_f((p[i] - sm) * sr);
}

// 3x3x3 SAME conv, planar NCDHW (N=1); bias only.
__global__ void k_conv3(const float* __restrict__ in, const float* __restrict__ w,
                        const float* __restrict__ b, float* __restrict__ out,
                        int Ci, int Co, int R) {
    int R3 = R * R * R;
    int idx = blockIdx.x * blockDim.x + threadIdx.x;
    if (idx >= Co * R3) return;
    int co = idx / R3;
    int rem = idx - co * R3;
    int z = rem / (R * R), y = (rem / R) % R, x = rem % R;
    float acc = b[co];
    for (int ci = 0; ci < Ci; ++ci) {
        const float* ip = in + (size_t)ci * R3;
        const float* wp = w + ((size_t)co * Ci + ci) * 27;
        for (int kz = -1; kz <= 1; ++kz) {
            int zz = z + kz; if (zz < 0 || zz >= R) continue;
            for (int ky = -1; ky <= 1; ++ky) {
                int yy = y + ky; if (yy < 0 || yy >= R) continue;
                for (int kx = -1; kx <= 1; ++kx) {
                    int xx = x + kx; if (xx < 0 || xx >= R) continue;
                    acc += ip[(zz * R + yy) * R + xx] * wp[(kz + 1) * 9 + (ky + 1) * 3 + (kx + 1)];
                }
            }
        }
    }
    out[idx] = acc;
}

// up_trilinear2: (C,S,S,S) -> (C,2S,2S,2S) with pos = od*(S-1)/(2S-1) per axis.
__global__ void k_up_tri(const float* __restrict__ in, float* __restrict__ out,
                         int C, int S) {
    int S2 = 2 * S, S2c = S2 * S2 * S2;
    int idx = blockIdx.x * blockDim.x + threadIdx.x;
    if (idx >= C * S2c) return;
    int c = idx / S2c;
    int rem = idx - c * S2c;
    int oz = rem / (S2 * S2), oy = (rem / S2) % S2, ox = rem % S2;
    float scale = (float)(S - 1) / (float)(S2 - 1);
    float pz = oz * scale; int z0 = (int)pz; if (z0 > S - 2) z0 = S - 2; float tz = pz - (float)z0;
    float py = oy * scale; int y0 = (int)py; if (y0 > S - 2) y0 = S - 2; float ty = py - (float)y0;
    float px = ox * scale; int x0 = (int)px; if (x0 > S - 2) x0 = S - 2; float tx = px - (float)x0;
    const float* ip = in + (size_t)c * S * S * S;
    int b0 = (z0 * S + y0) * S + x0;
    int SS = S * S;
    float c00 = ip[b0]          + (ip[b0 + 1]          - ip[b0])          * tx;
    float c01 = ip[b0 + S]      + (ip[b0 + S + 1]      - ip[b0 + S])      * tx;
    float c10 = ip[b0 + SS]     + (ip[b0 + SS + 1]     - ip[b0 + SS])     * tx;
    float c11 = ip[b0 + SS + S] + (ip[b0 + SS + S + 1] - ip[b0 + SS + S]) * tx;
    float d0 = c00 + (c01 - c00) * ty;
    float d1 = c10 + (c11 - c10) * ty;
    out[idx] = d0 + (d1 - d0) * tz;
}

// ---------------------------------------------------------------------------
// Weight packing for WMMA B-fragments (16-bit B 32x16 per-lane layout, mirror
// of the documented A layout: lane = half*16 + n; halves j<8 -> K=h*8+j,
// j>=8 -> K=16+h*8+(j-8); zero-pad K beyond Cin).
// ---------------------------------------------------------------------------
__global__ void k_pack_w(const float* __restrict__ W, _Float16* __restrict__ Bp,
                         int Cout, int Cin, int ksteps, int ntiles) {
    int tid = blockIdx.x * blockDim.x + threadIdx.x;
    int total = ksteps * ntiles * 32;
    if (tid >= total) return;
    int lane = tid & 31;
    int nt = (tid >> 5) % ntiles;
    int kk = tid / (32 * ntiles);
    int n = nt * 16 + (lane & 15);
    int hh = lane >> 4;
    for (int j = 0; j < 16; ++j) {
        int k = kk * 32 + ((j < 8) ? (hh * 8 + j) : (16 + hh * 8 + (j - 8)));
        float v = (k < Cin && n < Cout) ? W[(size_t)n * Cin + k] : 0.f;
        Bp[(size_t)tid * 16 + j] = (_Float16)v;
    }
}

// ---------------------------------------------------------------------------
// Trilinear grid-sample of one volume into an LDS feature row (f16).
// ---------------------------------------------------------------------------
__device__ __forceinline__ void sample_vol(const float* __restrict__ vol, int C, int R,
                                           float gx, float gy, float gz,
                                           _Float16* __restrict__ dst) {
    float fx = (gx + 1.f) * 0.5f * (float)(R - 1);
    float fy = (gy + 1.f) * 0.5f * (float)(R - 1);
    float fz = (gz + 1.f) * 0.5f * (float)(R - 1);
    int x0 = (int)floorf(fx); x0 = x0 < 0 ? 0 : (x0 > R - 2 ? R - 2 : x0);
    int y0 = (int)floorf(fy); y0 = y0 < 0 ? 0 : (y0 > R - 2 ? R - 2 : y0);
    int z0 = (int)floorf(fz); z0 = z0 < 0 ? 0 : (z0 > R - 2 ? R - 2 : z0);
    float tx = fx - (float)x0, ty = fy - (float)y0, tz = fz - (float)z0;
    float w00 = (1.f - tz) * (1.f - ty), w01 = (1.f - tz) * ty;
    float w10 = tz * (1.f - ty),         w11 = tz * ty;
    float w000 = w00 * (1.f - tx), w001 = w00 * tx;
    float w010 = w01 * (1.f - tx), w011 = w01 * tx;
    float w100 = w10 * (1.f - tx), w101 = w10 * tx;
    float w110 = w11 * (1.f - tx), w111 = w11 * tx;
    int R2 = R * R;
    int base = (z0 * R + y0) * R + x0;
    for (int c = 0; c < C; ++c) {
        const float* v = vol + (size_t)c * R2 * R + base;
        float r = w000 * v[0]       + w001 * v[1]
                + w010 * v[R]       + w011 * v[R + 1]
                + w100 * v[R2]      + w101 * v[R2 + 1]
                + w110 * v[R2 + R]  + w111 * v[R2 + R + 1];
        dst[c] = (_Float16)r;
    }
}

// ---------------------------------------------------------------------------
// MLP layer 1: fused grid-sample + feature assembly + WMMA GEMM.
// Block = 128 threads (4 waves); 64 points/block; K=192 (164 padded), N=64.
// Output h1 planar [64][NP] f32 (pre-norm).
// ---------------------------------------------------------------------------
__global__ void __launch_bounds__(128) k_mlp1(
    const float* __restrict__ grid, const float* __restrict__ app,
    const float* __restrict__ f1, const float* __restrict__ f2,
    const float* __restrict__ f3, const float* __restrict__ f4,
    const _Float16* __restrict__ Bp, const float* __restrict__ bias,
    float* __restrict__ h1) {
    __shared__ _Float16 feat[64][192];   // 24 KB

    int t = threadIdx.x;
    int lp = t & 63, halfsel = t >> 6;
    int p = blockIdx.x * 64 + lp;
    float gx = grid[(size_t)p * 3 + 0];
    float gy = grid[(size_t)p * 3 + 1];
    float gz = grid[(size_t)p * 3 + 2];

    // Phase 1: two threads per point, ~80 channels each.
    if (halfsel == 0) {
        feat[lp][0] = (_Float16)gx;
        feat[lp][1] = (_Float16)gy;
        feat[lp][2] = (_Float16)gz;
        sample_vol(f1, 64, 4,  gx, gy, gz, &feat[lp][3]);     // ch 3..66
        sample_vol(f4, 16, 32, gx, gy, gz, &feat[lp][147]);   // ch 147..162
        feat[lp][163] = (_Float16)app[p];
        for (int c = 164; c < 192; ++c) feat[lp][c] = (_Float16)0.f;  // K pad
    } else {
        sample_vol(f2, 48, 8,  gx, gy, gz, &feat[lp][67]);    // ch 67..114
        sample_vol(f3, 32, 16, gx, gy, gz, &feat[lp][115]);   // ch 115..146
    }
    __syncthreads();

    // Phase 2: each wave owns one 16-point tile. EXEC all-ones (no divergence).
    int wave = t >> 5, lane = t & 31;
    int m = lane & 15, hh = lane >> 4;
    const _Float16* arow = &feat[wave * 16 + m][0];
    const v16h* Bv = (const v16h*)Bp;

    v8f acc[4] = {};
#pragma unroll
    for (int kk = 0; kk < 6; ++kk) {
        union { v16h v; v8h h[2]; } a;
        a.h[0] = *(const v8h*)(arow + kk * 32 + hh * 8);        // K = kk*32 + h*8 .. +7
        a.h[1] = *(const v8h*)(arow + kk * 32 + 16 + hh * 8);   // K = kk*32+16+h*8 ..
#pragma unroll
        for (int nt = 0; nt < 4; ++nt) {
            v16h b = Bv[(kk * 4 + nt) * 32 + lane];
            acc[nt] = __builtin_amdgcn_wmma_f32_16x16x32_f16(
                false, a.v, false, b, (short)0, acc[nt], false, false);
        }
    }

    int tileBase = blockIdx.x * 64 + wave * 16;
#pragma unroll
    for (int nt = 0; nt < 4; ++nt) {
        int n = nt * 16 + m;          // D layout: lane holds one column N
        float bb = bias[n];
#pragma unroll
        for (int r = 0; r < 8; ++r)   // rows M = r + hh*8
            h1[(size_t)n * NP + tileBase + hh * 8 + r] = acc[nt][r] + bb;
    }
}

// Per-channel mean / rsqrt(var+eps) over NP points (planar input).
__global__ void k_stats(const float* __restrict__ buf, float* __restrict__ mean,
                        float* __restrict__ rstd, int S) {
    int c = blockIdx.x, tid = threadIdx.x;
    __shared__ float ssum[256], ssq[256];
    const float* p = buf + (size_t)c * S;
    float s = 0.f, q = 0.f;
    for (int i = tid; i < S; i += 256) { float v = p[i]; s += v; q += v * v; }
    ssum[tid] = s; ssq[tid] = q;
    __syncthreads();
    for (int o = 128; o > 0; o >>= 1) {
        if (tid < o) { ssum[tid] += ssum[tid + o]; ssq[tid] += ssq[tid + o]; }
        __syncthreads();
    }
    if (tid == 0) {
        float m = ssum[0] / (float)S;
        float v = ssq[0] / (float)S - m * m;
        mean[c] = m;
        rstd[c] = rsqrtf(v + EPSI);
    }
}

// MLP layer 2: normalize+relu h1 on the fly into LDS f16, WMMA K=64, N=32.
__global__ void __launch_bounds__(128) k_mlp2(
    const float* __restrict__ h1, const float* __restrict__ mean1,
    const float* __restrict__ rstd1, const _Float16* __restrict__ Bp,
    const float* __restrict__ bias, float* __restrict__ h2) {
    __shared__ _Float16 feat[64][64];   // 8 KB

    int t = threadIdx.x;
    int pBase = blockIdx.x * 64;
    for (int i = t; i < 64 * 64; i += 128) {
        int c = i >> 6, lp = i & 63;    // consecutive threads -> consecutive points
        float v = (h1[(size_t)c * NP + pBase + lp] - mean1[c]) * rstd1[c];
        feat[lp][c] = (_Float16)relu_f(v);
    }
    __syncthreads();

    int wave = t >> 5, lane = t & 31;
    int m = lane & 15, hh = lane >> 4;
    const _Float16* arow = &feat[wave * 16 + m][0];
    const v16h* Bv = (const v16h*)Bp;

    v8f acc[2] = {};
#pragma unroll
    for (int kk = 0; kk < 2; ++kk) {
        union { v16h v; v8h h[2]; } a;
        a.h[0] = *(const v8h*)(arow + kk * 32 + hh * 8);
        a.h[1] = *(const v8h*)(arow + kk * 32 + 16 + hh * 8);
#pragma unroll
        for (int nt = 0; nt < 2; ++nt) {
            v16h b = Bv[(kk * 2 + nt) * 32 + lane];
            acc[nt] = __builtin_amdgcn_wmma_f32_16x16x32_f16(
                false, a.v, false, b, (short)0, acc[nt], false, false);
        }
    }

    int tileBase = pBase + wave * 16;
#pragma unroll
    for (int nt = 0; nt < 2; ++nt) {
        int n = nt * 16 + m;
        float bb = bias[n];
#pragma unroll
        for (int r = 0; r < 8; ++r)
            h2[(size_t)n * NP + tileBase + hh * 8 + r] = acc[nt][r] + bb;
    }
}

// Final 32 -> 4 projection (scalar; 128 MAC/point). Writes d_out planar (1,4,64,64,64).
__global__ void k_mlp3(const float* __restrict__ h2, const float* __restrict__ mean2,
                       const float* __restrict__ rstd2, const float* __restrict__ w3,
                       const float* __restrict__ b3, float* __restrict__ out) {
    int p = blockIdx.x * blockDim.x + threadIdx.x;
    if (p >= NP) return;
    float o0 = b3[0], o1 = b3[1], o2 = b3[2], o3 = b3[3];
    for (int k = 0; k < 32; ++k) {
        float v = relu_f((h2[(size_t)k * NP + p] - mean2[k]) * rstd2[k]);
        o0 += w3[k] * v;
        o1 += w3[32 + k] * v;
        o2 += w3[64 + k] * v;
        o3 += w3[96 + k] * v;
    }
    out[0 * NP + p] = o0;
    out[1 * NP + p] = o1;
    out[2 * NP + p] = o2;
    out[3 * NP + p] = o3;
}

// ---------------------------------------------------------------------------
// Host orchestration
// ---------------------------------------------------------------------------
extern "C" void kernel_launch(void* const* d_in, const int* in_sizes, int n_in,
                              void* d_out, int out_size, void* d_ws, size_t ws_size,
                              hipStream_t stream) {
    (void)in_sizes; (void)n_in; (void)out_size; (void)ws_size;
    const float* lat   = (const float*)d_in[0];
    const float* grid  = (const float*)d_in[1];
    const float* app   = (const float*)d_in[2];
    const float* w_up1 = (const float*)d_in[3];  const float* b_up1 = (const float*)d_in[4];
    const float* w_up2 = (const float*)d_in[5];  const float* b_up2 = (const float*)d_in[6];
    const float* w_d1  = (const float*)d_in[7];  const float* b_d1  = (const float*)d_in[8];
    const float* w_d2a = (const float*)d_in[9];  const float* b_d2a = (const float*)d_in[10];
    const float* w_d2b = (const float*)d_in[11]; const float* b_d2b = (const float*)d_in[12];
    const float* w_d3a = (const float*)d_in[13]; const float* b_d3a = (const float*)d_in[14];
    const float* w_d3b = (const float*)d_in[15]; const float* b_d3b = (const float*)d_in[16];
    const float* w_d4a = (const float*)d_in[17]; const float* b_d4a = (const float*)d_in[18];
    const float* w_d4b = (const float*)d_in[19]; const float* b_d4b = (const float*)d_in[20];
    const float* w_m1  = (const float*)d_in[21]; const float* b_m1  = (const float*)d_in[22];
    const float* w_m2  = (const float*)d_in[23]; const float* b_m2  = (const float*)d_in[24];
    const float* w_m3  = (const float*)d_in[25]; const float* b_m3  = (const float*)d_in[26];

    float* ws = (float*)d_ws;
    size_t off = 0;
    auto alloc = [&](size_t n) { float* p = ws + off; off += n; return p; };
    float* x1 = alloc(128 * 8);
    float* x2 = alloc(64 * 64);
    float* f1 = alloc(64 * 64);
    float* u1 = alloc((size_t)64 * 512);
    float* t2 = alloc((size_t)48 * 512);
    float* f2 = alloc((size_t)48 * 512);
    float* u2 = alloc((size_t)48 * 4096);
    float* t3 = alloc((size_t)32 * 4096);
    float* f3 = alloc((size_t)32 * 4096);
    float* u3 = alloc((size_t)32 * 32768);
    float* t4 = alloc((size_t)16 * 32768);
    float* f4 = alloc((size_t)16 * 32768);
    float* h1 = alloc((size_t)64 * NP);
    float* h2 = alloc((size_t)32 * NP);
    float* mean1 = alloc(64); float* rstd1 = alloc(64);
    float* mean2 = alloc(32); float* rstd2 = alloc(32);
    _Float16* Bp1 = (_Float16*)(ws + off); off += (6 * 4 * 32 * 16) / 2;  // 12288 halves
    _Float16* Bp2 = (_Float16*)(ws + off); off += (2 * 2 * 32 * 16) / 2;  // 2048 halves

    // ---- trunk ----
    k_up1<<<4, 256, 0, stream>>>(lat, w_up1, b_up1, x1);
    k_up2<<<16, 256, 0, stream>>>(x1, w_up2, b_up2, x2);
    k_inorm_relu<<<64, 256, 0, stream>>>(x2, 64);
    k_conv3<<<(64 * 64 + 255) / 256, 256, 0, stream>>>(x2, w_d1, b_d1, f1, 64, 64, 4);
    k_inorm_relu<<<64, 256, 0, stream>>>(f1, 64);
    k_up_tri<<<(64 * 512 + 255) / 256, 256, 0, stream>>>(f1, u1, 64, 4);
    k_conv3<<<(48 * 512 + 255) / 256, 256, 0, stream>>>(u1, w_d2a, b_d2a, t2, 64, 48, 8);
    k_inorm_relu<<<48, 256, 0, stream>>>(t2, 512);
    k_conv3<<<(48 * 512 + 255) / 256, 256, 0, stream>>>(t2, w_d2b, b_d2b, f2, 48, 48, 8);
    k_inorm_relu<<<48, 256, 0, stream>>>(f2, 512);
    k_up_tri<<<(48 * 4096 + 255) / 256, 256, 0, stream>>>(f2, u2, 48, 8);
    k_conv3<<<(32 * 4096 + 255) / 256, 256, 0, stream>>>(u2, w_d3a, b_d3a, t3, 48, 32, 16);
    k_inorm_relu<<<32, 256, 0, stream>>>(t3, 4096);
    k_conv3<<<(32 * 4096 + 255) / 256, 256, 0, stream>>>(t3, w_d3b, b_d3b, f3, 32, 32, 16);
    k_inorm_relu<<<32, 256, 0, stream>>>(f3, 4096);
    k_up_tri<<<(32 * 32768 + 255) / 256, 256, 0, stream>>>(f3, u3, 32, 16);
    k_conv3<<<(16 * 32768 + 255) / 256, 256, 0, stream>>>(u3, w_d4a, b_d4a, t4, 32, 16, 32);
    k_inorm_relu<<<16, 256, 0, stream>>>(t4, 32768);
    k_conv3<<<(16 * 32768 + 255) / 256, 256, 0, stream>>>(t4, w_d4b, b_d4b, f4, 16, 16, 32);
    k_inorm_relu<<<16, 256, 0, stream>>>(f4, 32768);

    // ---- pack MLP weights into WMMA B-fragment layout (f16) ----
    k_pack_w<<<3, 256, 0, stream>>>(w_m1, Bp1, 64, 164, 6, 4);  // 768 lanes
    k_pack_w<<<1, 128, 0, stream>>>(w_m2, Bp2, 32, 64, 2, 2);   // 128 lanes

    // ---- per-point MLP on the WMMA path ----
    k_mlp1<<<NP / 64, 128, 0, stream>>>(grid, app, f1, f2, f3, f4, Bp1, b_m1, h1);
    k_stats<<<64, 256, 0, stream>>>(h1, mean1, rstd1, NP);
    k_mlp2<<<NP / 64, 128, 0, stream>>>(h1, mean1, rstd1, Bp2, b_m2, h2);
    k_stats<<<32, 256, 0, stream>>>(h2, mean2, rstd2, NP);
    k_mlp3<<<NP / 256, 256, 0, stream>>>(h2, mean2, rstd2, w_m3, b_m3, (float*)d_out);
}
